// LSTMModel_41644002902357
// MI455X (gfx1250) — matrix operand order repaired and checked by reading, not compiled, not run on hardware
//
#include <hip/hip_runtime.h>
#include <hip/hip_bf16.h>

#define T_ROWS 8192
#define P_DIM  256
#define Q_DIM  256
#define J_STK  500
#define B_SEQ  32
#define H_DIM  512
#define NSEG_N 1024
#define G4H    2048   // 4*H

typedef __attribute__((ext_vector_type(16))) __bf16 v16bf;
typedef __attribute__((ext_vector_type(8)))  float  v8f;

union Frag32B { uint4 q[2]; v16bf v; };

__device__ __forceinline__ unsigned short f2bf(float f) {
  unsigned int u = __float_as_uint(f);
  u += 0x7FFFu + ((u >> 16) & 1u);          // round-to-nearest-even
  return (unsigned short)(u >> 16);
}

// ---------------- prep: bf16 weights, fused bias, zero h/c ----------------
__global__ __launch_bounds__(256) void k_prep(
    const float* __restrict__ w_ih, const float* __restrict__ w_hh,
    const float* __restrict__ b_ih, const float* __restrict__ b_hh,
    unsigned short* __restrict__ Wih16, unsigned short* __restrict__ Whh16,
    float* __restrict__ bias, unsigned short* __restrict__ h16,
    float* __restrict__ cst) {
  int i = blockIdx.x * blockDim.x + threadIdx.x;
  if (i < G4H * P_DIM) Wih16[i] = f2bf(w_ih[i]);
  if (i < G4H * H_DIM) Whh16[i] = f2bf(w_hh[i]);
  if (i < G4H)         bias[i]  = b_ih[i] + b_hh[i];
  if (i < T_ROWS * H_DIM) { h16[i] = 0; cst[i] = 0.f; }
}

// ---------------- scores: W[t,b] = K[t,:] . S[:, idx[b]] ----------------
__global__ __launch_bounds__(256) void k_scores(
    const float* __restrict__ Km, const float* __restrict__ S,
    const int* __restrict__ idx, float* __restrict__ W) {
  __shared__ float ks[8][P_DIM];
  int b = threadIdx.x;            // 0..31
  int ty = threadIdx.y;           // 0..7
  int t0 = blockIdx.x * 8;
  int tid = ty * 32 + b;
  for (int i = tid; i < 8 * P_DIM; i += 256)
    ks[i / P_DIM][i % P_DIM] = Km[(size_t)(t0 + i / P_DIM) * P_DIM + (i % P_DIM)];
  __syncthreads();
  int col = idx[b];
  float acc = 0.f;
  #pragma unroll 4
  for (int k = 0; k < P_DIM; k++) acc += ks[ty][k] * S[(size_t)k * J_STK + col];
  W[(size_t)(t0 + ty) * B_SEQ + b] = acc;
}

// --------- segment softmax + weighted V sum -> Xb16 [B, NSEG, Q] (bf16) ---------
__global__ __launch_bounds__(256) void k_market(
    const float* __restrict__ W, const float* __restrict__ V,
    const int* __restrict__ dates, unsigned short* __restrict__ Xb16) {
  int seg = blockIdx.x, tid = threadIdx.x;   // tid = q (0..255)
  int r0 = dates[seg], r1 = dates[seg + 1];
  __shared__ float sm_m[B_SEQ], sm_rd[B_SEQ], sm_a[B_SEQ];
  if (tid < B_SEQ) {
    float m = -3.4e38f;
    for (int r = r0; r < r1; r++) m = fmaxf(m, W[(size_t)r * B_SEQ + tid]);
    float s = 0.f;
    for (int r = r0; r < r1; r++) s += expf(W[(size_t)r * B_SEQ + tid] - m);
    sm_m[tid] = m; sm_rd[tid] = 1.f / s;
  }
  __syncthreads();
  float acc[B_SEQ];
  #pragma unroll
  for (int b = 0; b < B_SEQ; b++) acc[b] = 0.f;
  for (int r = r0; r < r1; r++) {
    if (tid < B_SEQ) sm_a[tid] = expf(W[(size_t)r * B_SEQ + tid] - sm_m[tid]) * sm_rd[tid];
    __syncthreads();
    float vq = V[(size_t)r * Q_DIM + tid];
    #pragma unroll
    for (int b = 0; b < B_SEQ; b++) acc[b] += sm_a[b] * vq;
    __syncthreads();
  }
  #pragma unroll
  for (int b = 0; b < B_SEQ; b++)
    Xb16[((size_t)b * NSEG_N + seg) * Q_DIM + tid] = f2bf(acc[b]);
}

// ---------------- WMMA fragment loaders (ISA 7.12.2 layouts) ----------------
// A 16x32 bf16: lane g=l>>4, m=l&15; elem e<8 -> K=g*8+e ; e>=8 -> K=16+g*8+(e-8)
__device__ __forceinline__ v16bf load_a_frag(const unsigned short* base, int lda, int lane) {
  int g = lane >> 4, r = lane & 15;
  const unsigned short* p = base + (size_t)r * lda + g * 8;
  Frag32B u;
  u.q[0] = *(const uint4*)(p);
  u.q[1] = *(const uint4*)(p + 16);
  return u.v;
}
// B 32x16 bf16 from row-major [N,K] weights: lane n=l&15, elem e<8 -> K=g*8+e ; e>=8 -> K=16+g*8+(e-8)
__device__ __forceinline__ v16bf load_b_frag(const unsigned short* base, int ldb, int lane) {
  int g = lane >> 4, n = lane & 15;
  const unsigned short* p = base + (size_t)n * ldb + g * 8;
  Frag32B u;
  u.q[0] = *(const uint4*)(p);
  u.q[1] = *(const uint4*)(p + 16);
  return u.v;
}

#define WMMA_BF16(A, B, C) \
  __builtin_amdgcn_wmma_f32_16x16x32_bf16(false, (A), false, (B), (short)0, (C), false, false)

// -------- gates[T,4H] = h16 @ Whh16^T + (t<NSEG ? x16 @ Wih16^T : 0) + bias --------
__global__ __launch_bounds__(256) void k_gates(
    const unsigned short* __restrict__ h16,   // [T, 512] bf16
    const unsigned short* __restrict__ Whh16, // [2048, 512] bf16 (row-major N,K)
    const unsigned short* __restrict__ x16,   // [NSEG, 256] bf16 (this step)
    const unsigned short* __restrict__ Wih16, // [2048, 256] bf16
    const float* __restrict__ bias,           // [2048]
    float* __restrict__ gates) {              // [T, 2048]
  int tid = threadIdx.x;
  int lane = tid & 31;
  int w = tid >> 5;                  // 8 waves: 4 (M) x 2 (N)
  int wm = w & 3, wn = w >> 2;
  int tm = blockIdx.x * 128 + wm * 32;
  int tn = blockIdx.y * 64  + wn * 32;

  v8f c00 = {}, c01 = {}, c10 = {}, c11 = {};

  for (int kk = 0; kk < H_DIM; kk += 32) {
    v16bf a0 = load_a_frag(h16 + (size_t)tm * H_DIM + kk, H_DIM, lane);
    v16bf a1 = load_a_frag(h16 + (size_t)(tm + 16) * H_DIM + kk, H_DIM, lane);
    v16bf b0 = load_b_frag(Whh16 + (size_t)tn * H_DIM + kk, H_DIM, lane);
    v16bf b1 = load_b_frag(Whh16 + (size_t)(tn + 16) * H_DIM + kk, H_DIM, lane);
    c00 = WMMA_BF16(a0, b0, c00);
    c01 = WMMA_BF16(a0, b1, c01);
    c10 = WMMA_BF16(a1, b0, c10);
    c11 = WMMA_BF16(a1, b1, c11);
  }
  if (tm < NSEG_N) {  // x is zero for t >= NSEG; whole 128-row block agrees
    for (int kk = 0; kk < P_DIM; kk += 32) {
      v16bf a0 = load_a_frag(x16 + (size_t)tm * P_DIM + kk, P_DIM, lane);
      v16bf a1 = load_a_frag(x16 + (size_t)(tm + 16) * P_DIM + kk, P_DIM, lane);
      v16bf b0 = load_b_frag(Wih16 + (size_t)tn * P_DIM + kk, P_DIM, lane);
      v16bf b1 = load_b_frag(Wih16 + (size_t)(tn + 16) * P_DIM + kk, P_DIM, lane);
      c00 = WMMA_BF16(a0, b0, c00);
      c01 = WMMA_BF16(a0, b1, c01);
      c10 = WMMA_BF16(a1, b0, c10);
      c11 = WMMA_BF16(a1, b1, c11);
    }
  }
  // D layout: m = r + 8*(lane>>4), n = lane&15
  int g = lane >> 4, n = lane & 15;
  float bv0 = bias[tn + n], bv1 = bias[tn + 16 + n];
  #pragma unroll
  for (int r = 0; r < 8; r++) {
    int row0 = tm + 8 * g + r;
    int row1 = tm + 16 + 8 * g + r;
    gates[(size_t)row0 * G4H + tn + n]      = c00[r] + bv0;
    gates[(size_t)row0 * G4H + tn + 16 + n] = c01[r] + bv1;
    gates[(size_t)row1 * G4H + tn + n]      = c10[r] + bv0;
    gates[(size_t)row1 * G4H + tn + 16 + n] = c11[r] + bv1;
  }
}

// ------- pointwise LSTM cell + fused output projection (one block per t) -------
__global__ __launch_bounds__(512) void k_point(
    const float* __restrict__ gates, float* __restrict__ cst,
    unsigned short* __restrict__ h16, const float* __restrict__ w_lin,
    const float* __restrict__ b_lin, float* __restrict__ outb) {
  int t = blockIdx.x, j = threadIdx.x;       // j = hidden index 0..511
  const float* gr = gates + (size_t)t * G4H;
  float ig = 1.f / (1.f + expf(-gr[j]));
  float fg = 1.f / (1.f + expf(-gr[H_DIM + j]));
  float gg = tanhf(gr[2 * H_DIM + j]);
  float og = 1.f / (1.f + expf(-gr[3 * H_DIM + j]));
  size_t ci = (size_t)t * H_DIM + j;
  float cn = fg * cst[ci] + ig * gg;
  cst[ci] = cn;
  float h = og * tanhf(cn);
  h16[ci] = f2bf(h);
  __shared__ float red[512];
  red[j] = h * w_lin[j];
  __syncthreads();
  for (int s = 256; s > 0; s >>= 1) { if (j < s) red[j] += red[j + s]; __syncthreads(); }
  if (j == 0) outb[t] = red[0] + b_lin[0];
}

extern "C" void kernel_launch(void* const* d_in, const int* in_sizes, int n_in,
                              void* d_out, int out_size, void* d_ws, size_t ws_size,
                              hipStream_t stream) {
  (void)in_sizes; (void)n_in; (void)out_size; (void)ws_size;
  const float* Kin   = (const float*)d_in[0];
  const float* Vin   = (const float*)d_in[1];
  const float* Sin   = (const float*)d_in[2];
  const float* w_ih  = (const float*)d_in[3];
  const float* w_hh  = (const float*)d_in[4];
  const float* b_ih  = (const float*)d_in[5];
  const float* b_hh  = (const float*)d_in[6];
  const float* w_lin = (const float*)d_in[7];
  const float* b_lin = (const float*)d_in[8];
  const int*   idx   = (const int*)d_in[9];
  const int*   dates = (const int*)d_in[10];
  float* out = (float*)d_out;

  char* ws = (char*)d_ws;
  size_t off = 0;
  auto carve = [&](size_t bytes) -> char* {
    char* p = ws + off;
    off += (bytes + 255) & ~(size_t)255;
    return p;
  };
  unsigned short* Wih16 = (unsigned short*)carve((size_t)G4H * P_DIM * 2);     // 1 MB
  unsigned short* Whh16 = (unsigned short*)carve((size_t)G4H * H_DIM * 2);     // 2 MB
  float*          bias  = (float*)carve((size_t)G4H * 4);                      // 8 KB
  float*          Wsc   = (float*)carve((size_t)T_ROWS * B_SEQ * 4);           // 1 MB
  unsigned short* Xb16  = (unsigned short*)carve((size_t)B_SEQ * NSEG_N * Q_DIM * 2); // 16 MB
  unsigned short* h16   = (unsigned short*)carve((size_t)T_ROWS * H_DIM * 2);  // 8 MB
  float*          cst   = (float*)carve((size_t)T_ROWS * H_DIM * 4);           // 16 MB
  float*          gates = (float*)carve((size_t)T_ROWS * G4H * 4);             // 64 MB

  k_prep<<<(T_ROWS * H_DIM + 255) / 256, 256, 0, stream>>>(
      w_ih, w_hh, b_ih, b_hh, Wih16, Whh16, bias, h16, cst);

  dim3 sb(32, 8);
  k_scores<<<T_ROWS / 8, sb, 0, stream>>>(Kin, Sin, idx, Wsc);

  k_market<<<NSEG_N, 256, 0, stream>>>(Wsc, Vin, dates, Xb16);

  dim3 gg(T_ROWS / 128, G4H / 64);
  for (int b = 0; b < B_SEQ; b++) {
    k_gates<<<gg, 256, 0, stream>>>(h16, Whh16,
                                    Xb16 + (size_t)b * NSEG_N * Q_DIM,
                                    Wih16, bias, gates);
    k_point<<<T_ROWS, 512, 0, stream>>>(gates, cst, h16, w_lin, b_lin,
                                        out + (size_t)b * T_ROWS);
  }
}
